// EventModelDocV1_24283745092348
// MI455X (gfx1250) — matrix operand order repaired and checked by reading, not compiled, not run on hardware
//
#include <hip/hip_runtime.h>
#include <hip/hip_bf16.h>

// Bidirectional LSTM on gfx1250 (MI455X), wave32 + WMMA f16->f32.
//
// B=32 S=64 L=256 E=64 H=64 V=6000; N = B*S = 2048 sequences.
// One workgroup (4 waves / 128 threads) runs ONE direction for a tile of 16
// sequences over all 256 steps:
//   z = x_t[16x64] @ Wk[64x256] + h[16x64] @ Wr[64x256] + b
// = 16 x v_wmma_f32_16x16x32_f16 per step per wave (4 gate tiles x 4 K-chunks).
// Weights live in registers (pre-packed B fragments), ids live in LDS,
// h double-buffers through LDS (1 barrier/step), c stays in registers,
// gates are branchless v_exp/v_tanh/v_rcp.

typedef __attribute__((ext_vector_type(16))) _Float16 v16h;
typedef __attribute__((ext_vector_type(8)))  _Float16 v8h;
typedef __attribute__((ext_vector_type(8)))  float    v8f;

#define BB 32
#define SS 64
#define LL 256
#define EE 64
#define HH 64
#define VV 6000
#define NN (BB*SS)     // 2048 sequences
#define G4 (4*HH)      // 256 gate columns

// ---------------- fast branchless gate nonlinearities -------------------
__device__ __forceinline__ float fast_sigmoid(float x) {
  // 1/(1+2^(-x*log2e)); exp2(+-inf) saturates correctly through rcp.
  return __builtin_amdgcn_rcpf(1.f + __builtin_amdgcn_exp2f(-1.44269504088896340736f * x));
}
__device__ __forceinline__ float fast_tanh(float x) {
#if __has_builtin(__builtin_amdgcn_tanhf)
  return __builtin_amdgcn_tanhf(x);            // gfx1250 hardware v_tanh_f32
#else
  return 2.f * __builtin_amdgcn_rcpf(1.f + __builtin_amdgcn_exp2f(-2.88539008177792681472f * x)) - 1.f;
#endif
}

// ---------------- prep kernel 1: embed table f32 -> f16 ----------------
__global__ void embed_to_f16(const float* __restrict__ src,
                             _Float16* __restrict__ dst, int n) {
  int i = blockIdx.x * blockDim.x + threadIdx.x;
  if (i < n) dst[i] = (_Float16)src[i];
}

// ---------------- prep kernel 2: pack weights into WMMA B fragments -----
// dst[(((dir*2+mat)*16 + tile)*2 + kchunk)*32 + lane][elem]  (16 halves/lane)
// B-fragment (16-bit, 32x16): lane<16 -> col=lane,    elems = K 0..7,16..23 of chunk
//                             lane>=16-> col=lane-16, elems = K 8..15,24..31 of chunk
__global__ void pack_weights(const float* __restrict__ Wk_f,
                             const float* __restrict__ Wr_f,
                             const float* __restrict__ Wk_b,
                             const float* __restrict__ Wr_b,
                             _Float16* __restrict__ dst) {
  int i = blockIdx.x * blockDim.x + threadIdx.x;   // 2*2*16*2*32*16 = 65536
  if (i >= 65536) return;
  int e    = i & 15;
  int lane = (i >> 4) & 31;
  int c    = (i >> 9) & 1;
  int j    = (i >> 10) & 15;
  int mat  = (i >> 14) & 3;                        // dir*2 + {0:Wk,1:Wr}
  const float* W = (mat == 0) ? Wk_f : (mat == 1) ? Wr_f
                 : (mat == 2) ? Wk_b : Wr_b;
  int K   = 32 * c + 8 * (lane >> 4) + (e & 7) + 16 * (e >> 3);   // 0..63
  int col = 16 * j + (lane & 15);                  // 0..255
  dst[i] = (_Float16)W[K * G4 + col];
}

// ---------------- main kernel ------------------------------------------
union AFrag { v16h v; v8h h[2]; };

__global__ __launch_bounds__(128)
void bilstm_wmma_kernel(const int* __restrict__ ids,        // [N, L]
                        const _Float16* __restrict__ emb,   // [V, 64] f16
                        const _Float16* __restrict__ wpack, // packed B frags
                        const float* __restrict__ b_f,      // [256]
                        const float* __restrict__ b_b,      // [256]
                        float* __restrict__ out) {          // [N, L, 128]
  const int tile = blockIdx.x;          // 0..127  (16 sequences each)
  const int dir  = blockIdx.y;          // 0 = fwd, 1 = bwd
  const int w    = threadIdx.x >> 5;    // wave 0..3 -> hidden chunk 16w..16w+15
  const int lane = threadIdx.x & 31;
  const int nloc = lane & 15;           // A row / C column within 16x16 tile
  const int g    = lane >> 4;           // half-wave group
  const int n_base = tile * 16;

  // h double buffer: [buf][seq 16][hid 64] padded row stride 72 (b128-clean)
  __shared__ __align__(16) _Float16 hlds[2][16 * 72];
  // ids staged once: [seq 16][l 256] padded stride 257 (kills bank conflicts)
  __shared__ int ilds[16 * 257];

  for (int i = threadIdx.x; i < 16 * 256; i += 128) {
    int r = i >> 8, cc = i & 255;
    ilds[r * 257 + cc] = ids[(size_t)(n_base + r) * LL + cc];
  }
  for (int i = threadIdx.x; i < 2 * 16 * 72; i += 128)
    ((_Float16*)hlds)[i] = (_Float16)0.f;

  // ---- weight B-fragments: load once, keep in registers for all 256 steps
  v16h wk[4][2], wr[4][2];
#pragma unroll
  for (int q = 0; q < 4; ++q) {
    const int tj = 4 * q + w;           // gate tile: i->w, f->4+w, g->8+w, o->12+w
#pragma unroll
    for (int c = 0; c < 2; ++c) {
      size_t bk = ((((size_t)(dir * 2 + 0) * 16 + tj) * 2 + c) * 32 + lane) * 16;
      size_t br = ((((size_t)(dir * 2 + 1) * 16 + tj) * 2 + c) * 32 + lane) * 16;
      wk[q][c] = *(const v16h*)(wpack + bk);
      wr[q][c] = *(const v16h*)(wpack + br);
    }
  }

  const float* bias = dir ? b_b : b_f;
  float bv[4];
#pragma unroll
  for (int q = 0; q < 4; ++q) bv[q] = bias[64 * q + 16 * w + nloc];

  float creg[8];                        // cell state, C layout (col nloc, 8 rows)
#pragma unroll
  for (int r = 0; r < 8; ++r) creg[r] = 0.f;

  const int hid = 16 * w + nloc;        // this lane's hidden unit (C column)
  __syncthreads();                      // ilds / hlds ready

  // ---- prefetch x A-fragments for t = 0
  // 16-bit A 16x32 layout: elems0-7 <- K = 32c+8g.., elems8-15 <- K = 32c+16+8g..
  AFrag xa[2], xn[2];
  {
    const int l0 = dir ? (LL - 1) : 0;
    const int id0 = ilds[nloc * 257 + l0];
    const _Float16* xr = emb + (size_t)id0 * EE;
#pragma unroll
    for (int c = 0; c < 2; ++c) {
      xa[c].h[0] = *(const v8h*)(xr + 32 * c + 8 * g);
      xa[c].h[1] = *(const v8h*)(xr + 32 * c + 16 + 8 * g);
    }
  }

  int pb = 0;                           // buffer holding h_{t-1}
#pragma unroll 2
  for (int t = 0; t < LL; ++t) {
    const int l = dir ? (LL - 1 - t) : t;

    // ---- h_{t-1} A-fragments from LDS buffer pb
    AFrag ha[2];
    const _Float16* hrow = hlds[pb] + nloc * 72;
#pragma unroll
    for (int c = 0; c < 2; ++c) {
      ha[c].h[0] = *(const v8h*)(hrow + 32 * c + 8 * g);
      ha[c].h[1] = *(const v8h*)(hrow + 32 * c + 16 + 8 * g);
    }

    // ---- prefetch x for t+1 (overlaps with WMMA + gate math below)
    if (t + 1 < LL) {
      const int ln = dir ? (LL - 2 - t) : (t + 1);
      const int idn = ilds[nloc * 257 + ln];
      const _Float16* xr = emb + (size_t)idn * EE;
#pragma unroll
      for (int c = 0; c < 2; ++c) {
        xn[c].h[0] = *(const v8h*)(xr + 32 * c + 8 * g);
        xn[c].h[1] = *(const v8h*)(xr + 32 * c + 16 + 8 * g);
      }
    }

    // ---- z = x@Wk + h@Wr : 16 WMMAs, f32 accumulate
    v8f acc[4];
#pragma unroll
    for (int q = 0; q < 4; ++q) {
      v8f a = {};
      a = __builtin_amdgcn_wmma_f32_16x16x32_f16(false, xa[0].v, false, wk[q][0], (short)0, a, false, false);
      a = __builtin_amdgcn_wmma_f32_16x16x32_f16(false, xa[1].v, false, wk[q][1], (short)0, a, false, false);
      a = __builtin_amdgcn_wmma_f32_16x16x32_f16(false, ha[0].v, false, wr[q][0], (short)0, a, false, false);
      a = __builtin_amdgcn_wmma_f32_16x16x32_f16(false, ha[1].v, false, wr[q][1], (short)0, a, false, false);
      acc[q] = a;
    }

    // ---- gates (Keras i,f,g,o), branchless; write h_t into buffer 1-pb
    float hnew[8];
    _Float16* hwr = hlds[1 - pb];
#pragma unroll
    for (int r = 0; r < 8; ++r) {
      float iv = fast_sigmoid(acc[0][r] + bv[0]);
      float fv = fast_sigmoid(acc[1][r] + bv[1]);
      float gv = fast_tanh   (acc[2][r] + bv[2]);
      float ov = fast_sigmoid(acc[3][r] + bv[3]);
      float cn = fv * creg[r] + iv * gv;
      creg[r] = cn;
      float hv = ov * fast_tanh(cn);
      hnew[r] = hv;
      hwr[(8 * g + r) * 72 + hid] = (_Float16)hv;   // C->A transpose via LDS
    }

    // ---- output [n, l, dir*64 + hid] f32; lanes 0..15 cover 16 consecutive
    // hidden units -> 64B coalesced runs per row
#pragma unroll
    for (int r = 0; r < 8; ++r) {
      const int m = 8 * g + r;
      size_t oidx = ((size_t)(n_base + m) * LL + l) * (2 * HH) + dir * HH + hid;
      out[oidx] = hnew[r];
    }

    __syncthreads();                    // h_t visible before next step's reads
    pb ^= 1;
#pragma unroll
    for (int c = 0; c < 2; ++c) xa[c] = xn[c];
  }
}

extern "C" void kernel_launch(void* const* d_in, const int* in_sizes, int n_in,
                              void* d_out, int out_size, void* d_ws, size_t ws_size,
                              hipStream_t stream) {
  const int*   ids  = (const int*)d_in[0];
  const float* emb  = (const float*)d_in[1];
  const float* Wk_f = (const float*)d_in[2];
  const float* Wr_f = (const float*)d_in[3];
  const float* b_f  = (const float*)d_in[4];
  const float* Wk_b = (const float*)d_in[5];
  const float* Wr_b = (const float*)d_in[6];
  const float* b_b  = (const float*)d_in[7];
  float* out = (float*)d_out;

  // Workspace: [0, 768000) embed f16; [768000, +131072) packed weight frags.
  _Float16* embF  = (_Float16*)d_ws;
  _Float16* wpack = (_Float16*)((char*)d_ws + 768000);

  const int nEmb = VV * EE;                       // 384000
  embed_to_f16<<<(nEmb + 255) / 256, 256, 0, stream>>>(emb, embF, nEmb);
  pack_weights<<<65536 / 256, 256, 0, stream>>>(Wk_f, Wr_f, Wk_b, Wr_b, wpack);

  dim3 grid(NN / 16, 2);                          // 128 sequence tiles x 2 dirs
  bilstm_wmma_kernel<<<grid, 128, 0, stream>>>(ids, embF, wpack, b_f, b_b, out);
}
